// Correlation_55946243998251
// MI455X (gfx1250) — matrix operand order repaired and checked by reading, not compiled, not run on hardware
//
#include <hip/hip_runtime.h>

typedef __attribute__((ext_vector_type(16))) _Float16 v16h;
typedef __attribute__((ext_vector_type(8)))  float    v8f;

#define BATCH   8
#define CHN     196
#define HEIGHT  128
#define WIDTH   224
#define HW      (HEIGHT*WIDTH)
#define PADR    4
#define ND      9            // displacements per axis
#define NDISP   81
#define HT      4            // output rows per block
#define WT      16           // output pixels (w) per block
#define KC      32           // channel chunk (WMMA K)
#define NKC     7            // ceil(196/32)
#define NW      9            // waves per block (one per dh)
#define NTHREADS (NW*32)
#define BROWS   (HT + ND - 1)   // 12 f2 rows staged
#define BCOLS   (WT + 8)        // 24 f2 cols staged
#define KPAD    40              // padded k-dim (bank spread, keeps 16B align)

__device__ __forceinline__ int refl(int i, int n) {
  i = (i < 0) ? -i : i;
  return (i >= n) ? (2 * n - 2 - i) : i;
}

union FragH  { float4 f4[2]; v16h v; };
union Tile32 { _Float16 h[32]; float4 f4[4]; };

__global__ __launch_bounds__(NTHREADS)
void corr_wmma_kernel(const float* __restrict__ in1,
                      const float* __restrict__ in2,
                      float* __restrict__ out) {
  __shared__ _Float16 sA[HT][WT][KPAD];       // f1 tile, k-contiguous
  __shared__ _Float16 sB[BROWS][BCOLS][KPAD]; // f2 rows, k-contiguous
  __shared__ float    sO[HT][NDISP][WT];      // output staging

  const int w0   = blockIdx.x * WT;
  const int h0   = blockIdx.y * HT;
  const int b    = blockIdx.z;
  const int tid  = threadIdx.x;
  const int wid  = tid >> 5;          // wave id == dh
  const int lane = tid & 31;
  const int sel  = lane >> 4;         // half-wave selector
  const int idx  = lane & 15;

  // ---- loop-invariant staging geometry ----------------------------------
  // B: 288 threads = 12 rows x 24 cols, fixed (rB, nB) per thread
  const int nB   = tid % BCOLS;
  const int rB   = tid / BCOLS;                 // [0,12)
  const int hinB = refl(h0 + rB - PADR, HEIGHT);
  const int winB = refl(w0 + nB - PADR, WIDTH);
  const float* __restrict__ baseB =
      in2 + (size_t)b * CHN * HW + (size_t)hinB * WIDTH + winB;
  // A: t -> (m = t&15, g = t>>4); walk flattened (hl,k) rows with stride 18
  const int mA = tid & 15;
  const int gA = tid >> 4;                      // [0,18)
  const float* __restrict__ baseA =
      in1 + (size_t)b * CHN * HW + (size_t)h0 * WIDTH + w0 + mA;

  v8f acc[HT][2];
  for (int hl = 0; hl < HT; ++hl)
    for (int t = 0; t < 2; ++t)
      for (int i = 0; i < 8; ++i) acc[hl][t][i] = 0.0f;

  for (int kc = 0; kc < NKC; ++kc) {
    const int c0 = kc * KC;
    __syncthreads();

    // ---- stage B: fixed (rB,nB); walk k, write 64 contiguous bytes ------
    {
      Tile32 tb;
      const float* p = baseB + (size_t)c0 * HW;
#pragma unroll
      for (int k = 0; k < KC; ++k) {
        float v = 0.0f;
        if (c0 + k < CHN) v = p[(size_t)k * HW];
        tb.h[k] = (_Float16)v;
      }
      float4* dst = (float4*)&sB[rB][nB][0];
#pragma unroll
      for (int i = 0; i < 4; ++i) dst[i] = tb.f4[i];
    }
    // ---- stage A: rows R = gA + 18*i over flattened (hl,k) --------------
    for (int R = gA; R < HT * KC; R += 18) {
      int hl = R >> 5;
      int k  = R & 31;
      float v = 0.0f;
      if (c0 + k < CHN) v = baseA[(size_t)(c0 + k) * HW + hl * WIDTH];
      sA[hl][mA][k] = (_Float16)v;
    }
    __syncthreads();

    const int n1 = 16 + (idx < 8 ? idx : 7);  // tile-1 col (cols >=8 unused)
#pragma unroll
    for (int hl = 0; hl < HT; ++hl) {
      FragH fa, fb0, fb1;
      // A frag: lane row M=idx; K run = 16*sel .. 16*sel+15
      const float4* pa = (const float4*)&sA[hl][idx][sel * 16];
      fa.f4[0] = pa[0];
      fa.f4[1] = pa[1];
      // B frag: lane col N=idx; K run = 16*sel .. 16*sel+15
      const float4* pb0 = (const float4*)&sB[hl + wid][idx][sel * 16];
      fb0.f4[0] = pb0[0];
      fb0.f4[1] = pb0[1];
      const float4* pb1 = (const float4*)&sB[hl + wid][n1][sel * 16];
      fb1.f4[0] = pb1[0];
      fb1.f4[1] = pb1[1];
      acc[hl][0] = __builtin_amdgcn_wmma_f32_16x16x32_f16(
          false, fa.v, false, fb0.v, (short)0, acc[hl][0], false, false);
      acc[hl][1] = __builtin_amdgcn_wmma_f32_16x16x32_f16(
          false, fa.v, false, fb1.v, (short)0, acc[hl][1], false, false);
    }
  }

  __syncthreads();
  // ---- scatter banded GEMM results into LDS out-tile --------------------
  const float scale = 1.0f / (float)CHN;
#pragma unroll
  for (int hl = 0; hl < HT; ++hl) {
#pragma unroll
    for (int v = 0; v < 8; ++v) {
      int m = v + 8 * sel;          // pixel index within tile
      int n = idx;                  // B column
      int dw0 = n - m;              // tile 0: dw = n - m
      if (dw0 >= 0 && dw0 <= 8)
        sO[hl][wid * ND + dw0][m] = acc[hl][0][v] * scale;
      int dw1 = n + 16 - m;         // tile 1: dw = n + 16 - m
      if (dw1 >= 0 && dw1 <= 8)
        sO[hl][wid * ND + dw1][m] = acc[hl][1][v] * scale;
    }
  }
  __syncthreads();
  // ---- coalesced store: out[b, d, h0+hl, w0+m]; shift-only decode -------
  for (int hl = 0; hl < HT; ++hl) {
    float* __restrict__ orow =
        out + ((size_t)b * NDISP * HEIGHT + (size_t)(h0 + hl)) * WIDTH + w0;
    for (int e = tid; e < NDISP * WT; e += NTHREADS) {
      int d = e >> 4;
      int m = e & 15;
      orow[(size_t)d * HW + m] = sO[hl][d][m];
    }
  }
}

extern "C" void kernel_launch(void* const* d_in, const int* in_sizes, int n_in,
                              void* d_out, int out_size, void* d_ws, size_t ws_size,
                              hipStream_t stream) {
  const float* in1 = (const float*)d_in[0];
  const float* in2 = (const float*)d_in[1];
  float* out = (float*)d_out;
  dim3 grid(WIDTH / WT, HEIGHT / HT, BATCH);  // 14 x 32 x 8
  dim3 block(NTHREADS);                       // 288 = 9 waves (one per dh)
  corr_wmma_kernel<<<grid, block, 0, stream>>>(in1, in2, out);
}
